// AttentionRNN_15350213116309
// MI455X (gfx1250) — matrix operand order repaired and checked by reading, not compiled, not run on hardware
//
#include <hip/hip_runtime.h>

typedef __attribute__((ext_vector_type(16))) _Float16 v16h;
typedef __attribute__((ext_vector_type(8)))  float    v8f;

#define WMMA_F16(a, b, c) \
  __builtin_amdgcn_wmma_f32_16x16x32_f16(false, (a), false, (b), (short)0, (c), false, false)

// CDNA5 has V_TANH_F32 (TRANS op). Use the builtin when the toolchain exposes
// it; otherwise fall back to OCML tanhf.
#if defined(__has_builtin)
#if __has_builtin(__builtin_amdgcn_tanhf)
#define FAST_TANH(x) __builtin_amdgcn_tanhf(x)
#endif
#endif
#ifndef FAST_TANH
#define FAST_TANH(x) tanhf(x)
#endif

// A-fragment (16-bit 16x32) flat offset inside a [kb][32 lanes][16 halves] block:
// lane l holds row m=l&15; half i -> K = (i>>3)*16 + (l>>4)*8 + (i&7)
__device__ __forceinline__ int afrag_off(int m, int K) {
  int kb   = K >> 5;
  int r    = K & 31;
  int i    = ((r >> 4) << 3) | (r & 7);
  int lane = m | (((r >> 3) & 1) << 4);
  return ((kb << 5) + lane) * 16 + i;
}

// ---------------- prep kernels ----------------

// Convert row-major f32 weight W[Nout][Kin] into f16 B-fragment order:
// dst[((tile*kblocks+kb)*32+lane)*16+i] = W[tile*16+(lane&15)][kb*32+((i>>3)<<4)+((lane>>4)<<3)+(i&7)]
__global__ __launch_bounds__(256) void fragB_fill(const float* __restrict__ src,
                                                  _Float16* __restrict__ dst,
                                                  int Nout, int Kin, int kblocks) {
  int idx  = blockIdx.x * 256 + threadIdx.x;
  int i    = idx & 15;
  int lane = (idx >> 4) & 31;
  int tk   = idx >> 9;
  int kb   = tk % kblocks;
  int tile = tk / kblocks;
  int n = tile * 16 + (lane & 15);
  int K = kb * 32 + ((i >> 3) << 4) + ((lane >> 4) << 3) + (i & 7);
  float v = (n < Nout && K < Kin) ? src[n * Kin + K] : 0.f;
  dst[idx] = (_Float16)v;
}

// Embedding gather into A-fragment order per timestep: embA[t][kb(2)][32][16]
__global__ __launch_bounds__(256) void embA_fill(const int* __restrict__ x,
                                                 const float* __restrict__ embed,
                                                 _Float16* __restrict__ dst) {
  int idx  = blockIdx.x * 256 + threadIdx.x;   // total 256*2*512 = 262144
  int i    = idx & 15;
  int lane = (idx >> 4) & 31;
  int kb   = (idx >> 9) & 1;
  int t    = idx >> 10;
  int m = lane & 15;
  int K = kb * 32 + ((i >> 3) << 4) + ((lane >> 4) << 3) + (i & 7);
  float v = 0.f;
  if (m < 8 && K < 64) {
    int tok = x[m * 257 + t];                  // inp = x[:, :-1], SEQ stride 257
    v = embed[tok * 64 + K];
  }
  dst[idx] = (_Float16)v;
}

__global__ void bias_sum(const float* __restrict__ a, const float* __restrict__ b,
                         const float* __restrict__ c, const float* __restrict__ d,
                         float* __restrict__ s0, float* __restrict__ s1) {
  int i = threadIdx.x;   // 256
  s0[i] = a[i] + b[i];
  s1[i] = c[i] + d[i];
}

// ---------------- RNN (single workgroup, 16 waves, sequential over T) ----------------

__global__ __launch_bounds__(512) void rnn_kernel(
    const _Float16* __restrict__ embA,
    const _Float16* __restrict__ Wih0B, const _Float16* __restrict__ Whh0B,
    const _Float16* __restrict__ Wih1B, const _Float16* __restrict__ Whh1B,
    const float* __restrict__ bsum0, const float* __restrict__ bsum1,
    float* __restrict__ Hs,            // [8][256][256] f32
    _Float16* __restrict__ concatA)    // [128 tiles][16 kb][32][16] f16 A-frags (K=0..255 = Hs)
{
  __shared__ _Float16 h0A[8 * 32 * 16];
  __shared__ _Float16 h1A[8 * 32 * 16];
  const int tid  = threadIdx.x;
  const int w    = tid >> 5;
  const int lane = tid & 31;
  for (int idx = tid; idx < 8 * 32 * 16; idx += 512) {
    h0A[idx] = (_Float16)0.f;
    h1A[idx] = (_Float16)0.f;
  }
  // warm the recurrent weights into cache (reused 256x)
  for (int p = tid * 128; p < 16 * 8 * 512; p += 512 * 128) {
    __builtin_prefetch(Whh0B + p, 0, 1);
    __builtin_prefetch(Wih1B + p, 0, 1);
    __builtin_prefetch(Whh1B + p, 0, 1);
  }
  __syncthreads();

  const int nt    = w;                       // each wave owns one 16-wide N tile
  const int ncol  = nt * 16 + (lane & 15);
  const float b0  = bsum0[ncol];
  const float b1  = bsum1[ncol];
  const int mrow  = (lane >> 4) << 3;

  for (int t = 0; t < 256; ++t) {
    // ---- layer 0: h0 = tanh(e_t@Wih0^T + h0@Whh0^T + bias0) ----
    v8f c0; for (int j = 0; j < 8; ++j) c0[j] = b0;
    #pragma unroll
    for (int kb = 0; kb < 2; ++kb) {
      v16h a = *(const v16h*)(embA  + ((t  * 2 + kb) * 32 + lane) * 16);
      v16h b = *(const v16h*)(Wih0B + ((nt * 2 + kb) * 32 + lane) * 16);
      c0 = WMMA_F16(a, b, c0);
    }
    #pragma unroll
    for (int kb = 0; kb < 8; ++kb) {
      v16h a = *(const v16h*)(h0A + ((kb << 5) + lane) * 16);
      v16h b = *(const v16h*)(Whh0B + ((nt * 8 + kb) * 32 + lane) * 16);
      c0 = WMMA_F16(a, b, c0);
    }
    __syncthreads();                         // everyone done reading h0A(prev)
    float h0v[8];
    for (int j = 0; j < 8; ++j) h0v[j] = FAST_TANH(c0[j]);
    for (int j = 0; j < 8; ++j)
      h0A[afrag_off(j + mrow, ncol)] = (_Float16)h0v[j];
    __syncthreads();                         // h0A(new) visible

    // ---- layer 1: h1 = tanh(h0@Wih1^T + h1@Whh1^T + bias1) ----
    v8f c1; for (int j = 0; j < 8; ++j) c1[j] = b1;
    #pragma unroll
    for (int kb = 0; kb < 8; ++kb) {
      v16h a = *(const v16h*)(h0A + ((kb << 5) + lane) * 16);
      v16h b = *(const v16h*)(Wih1B + ((nt * 8 + kb) * 32 + lane) * 16);
      c1 = WMMA_F16(a, b, c1);
    }
    #pragma unroll
    for (int kb = 0; kb < 8; ++kb) {
      v16h a = *(const v16h*)(h1A + ((kb << 5) + lane) * 16);
      v16h b = *(const v16h*)(Whh1B + ((nt * 8 + kb) * 32 + lane) * 16);
      c1 = WMMA_F16(a, b, c1);
    }
    __syncthreads();                         // everyone done reading h1A(prev)
    float h1v[8];
    for (int j = 0; j < 8; ++j) h1v[j] = FAST_TANH(c1[j]);
    for (int j = 0; j < 8; ++j)
      h1A[afrag_off(j + mrow, ncol)] = (_Float16)h1v[j];
    // publish Hs: f32 plain + f16 A-fragments (rows flattened b*256+t)
    for (int j = 0; j < 8; ++j) {
      int m = j + mrow;                      // batch index (8..15 are padding)
      if (m < 8) {
        Hs[(m * 256 + t) * 256 + ncol] = h1v[j];
        int tile = m * 16 + (t >> 4);
        concatA[tile * 8192 + afrag_off(t & 15, ncol)] = (_Float16)h1v[j];
      }
    }
    __syncthreads();
  }
}

// ---------------- q = Hs@Wq^T, kT[b][h][s] = (Hs@Wk^T)[b][s][h] ----------------

__global__ __launch_bounds__(256) void qk_kernel(
    const _Float16* __restrict__ concatA,   // K=0..255 region = Hs A-frags
    const _Float16* __restrict__ WqB, const _Float16* __restrict__ WkB,
    float* __restrict__ q, float* __restrict__ kT)
{
  int gw    = (blockIdx.x * 256 + threadIdx.x) >> 5;  // 0..4095
  int lane  = threadIdx.x & 31;
  int which = gw & 1;                                 // 0=q, 1=k
  int ntile = (gw >> 1) & 15;
  int mtile = gw >> 5;                                // 0..127
  const _Float16* WB = which ? WkB : WqB;
  v8f c; for (int j = 0; j < 8; ++j) c[j] = 0.f;
  #pragma unroll
  for (int kb = 0; kb < 8; ++kb) {
    v16h a = *(const v16h*)(concatA + mtile * 8192 + ((kb << 5) + lane) * 16);
    v16h b = *(const v16h*)(WB + ((ntile * 8 + kb) * 32 + lane) * 16);
    c = WMMA_F16(a, b, c);
  }
  int n    = ntile * 16 + (lane & 15);
  int mrow = (lane >> 4) << 3;
  for (int j = 0; j < 8; ++j) {
    int row = mtile * 16 + j + mrow;                  // = b*256 + t
    if (which == 0) {
      q[row * 256 + n] = c[j];
    } else {
      int bb = row >> 8, s = row & 255;
      kT[(bb * 256 + n) * 256 + s] = c[j];            // transposed for coalesced attn
    }
  }
}

// ---------------- additive attention: scores -> softmax -> ctx ----------------

__global__ __launch_bounds__(256) void attn_kernel(
    const float* __restrict__ q, const float* __restrict__ kT,
    const float* __restrict__ Hs, const float* __restrict__ vvec,
    _Float16* __restrict__ concatA)
{
  int b    = blockIdx.x >> 8;
  int t    = blockIdx.x & 255;
  int tid  = threadIdx.x;
  int tile = b * 16 + (t >> 4);
  int rowm = t & 15;
  if (t == 0) {                                       // ctx masked to zero at t=0
    concatA[tile * 8192 + afrag_off(rowm, 256 + tid)] = (_Float16)0.f;
    return;
  }
  __shared__ float qs[256], vs[256], red[256], al[256];
  qs[tid] = q[(b * 256 + t) * 256 + tid];
  vs[tid] = vvec[tid];
  __syncthreads();

  int s = tid;
  float score = -1e30f;
  if (s < t) {                                        // strict causal mask
    const float* kp = kT + (size_t)b * 65536 + s;     // kT[b][h][s], coalesced in s
    float acc = 0.f;
    #pragma unroll 4
    for (int h = 0; h < 256; ++h)
      acc = fmaf(vs[h], FAST_TANH(qs[h] + kp[h * 256]), acc);
    score = acc;
  }
  // softmax over s < t
  red[tid] = score; __syncthreads();
  for (int off = 128; off > 0; off >>= 1) {
    if (tid < off) red[tid] = fmaxf(red[tid], red[tid + off]);
    __syncthreads();
  }
  float mx = red[0]; __syncthreads();
  float ex = (s < t) ? __expf(score - mx) : 0.f;
  red[tid] = ex; __syncthreads();
  for (int off = 128; off > 0; off >>= 1) {
    if (tid < off) red[tid] += red[tid + off];
    __syncthreads();
  }
  float inv = 1.f / red[0];
  al[tid] = ex * inv; __syncthreads();

  // ctx[h] = sum_{s<t} alpha[s] * Hs[b][s][h]  (coalesced in h)
  float acc = 0.f;
  const float* hp = Hs + (size_t)b * 65536 + tid;
  for (int s2 = 0; s2 < t; ++s2) acc = fmaf(al[s2], hp[s2 * 256], acc);
  concatA[tile * 8192 + afrag_off(rowm, 256 + tid)] = (_Float16)acc;
}

// ---------------- FC head: out = concat(Hs,ctx) @ Wfc^T + bfc ----------------

__global__ __launch_bounds__(256) void fc_kernel(
    const _Float16* __restrict__ concatA, const _Float16* __restrict__ WfcB,
    const float* __restrict__ bfc, float* __restrict__ out)
{
  int gw    = (blockIdx.x * 256 + threadIdx.x) >> 5;  // 0..255
  int lane  = threadIdx.x & 31;
  int ntile = gw & 1;                                 // vocab 29 -> 2 N tiles (padded)
  int mtile = gw >> 1;                                // 0..127
  int n = ntile * 16 + (lane & 15);
  float bv = (n < 29) ? bfc[n] : 0.f;
  v8f c; for (int j = 0; j < 8; ++j) c[j] = bv;
  #pragma unroll
  for (int kb = 0; kb < 16; ++kb) {
    v16h a = *(const v16h*)(concatA + mtile * 8192 + ((kb << 5) + lane) * 16);
    v16h b = *(const v16h*)(WfcB + ((ntile * 16 + kb) * 32 + lane) * 16);
    c = WMMA_F16(a, b, c);
  }
  int mrow = (lane >> 4) << 3;
  if (n < 29) {
    for (int j = 0; j < 8; ++j) {
      int row = mtile * 16 + j + mrow;                // = b*256 + t
      out[row * 29 + n] = c[j];
    }
  }
}

// ---------------- host ----------------

extern "C" void kernel_launch(void* const* d_in, const int* in_sizes, int n_in,
                              void* d_out, int out_size, void* d_ws, size_t ws_size,
                              hipStream_t stream) {
  const int*   x     = (const int*)d_in[0];
  const float* embed = (const float*)d_in[1];
  const float* Wih0  = (const float*)d_in[2];
  const float* bih0  = (const float*)d_in[3];
  const float* Whh0  = (const float*)d_in[4];
  const float* bhh0  = (const float*)d_in[5];
  const float* Wih1  = (const float*)d_in[6];
  const float* bih1  = (const float*)d_in[7];
  const float* Whh1  = (const float*)d_in[8];
  const float* bhh1  = (const float*)d_in[9];
  const float* Wq    = (const float*)d_in[10];
  const float* Wk    = (const float*)d_in[11];
  const float* vvec  = (const float*)d_in[12];
  const float* Wfc   = (const float*)d_in[13];
  const float* bfc   = (const float*)d_in[14];
  float* out = (float*)d_out;

  char* ws = (char*)d_ws;
  size_t off = 0;
  auto alloc = [&](size_t bytes) -> void* {
    off = (off + 255) & ~(size_t)255;
    void* p = ws + off;
    off += bytes;
    return p;
  };
  _Float16* Wih0B  = (_Float16*)alloc((size_t)16 * 2  * 512 * 2);   // 32 KB
  _Float16* Whh0B  = (_Float16*)alloc((size_t)16 * 8  * 512 * 2);   // 128 KB
  _Float16* Wih1B  = (_Float16*)alloc((size_t)16 * 8  * 512 * 2);
  _Float16* Whh1B  = (_Float16*)alloc((size_t)16 * 8  * 512 * 2);
  _Float16* WqB    = (_Float16*)alloc((size_t)16 * 8  * 512 * 2);
  _Float16* WkB    = (_Float16*)alloc((size_t)16 * 8  * 512 * 2);
  _Float16* WfcB   = (_Float16*)alloc((size_t)2  * 16 * 512 * 2);   // 32 KB
  float*    bsum0  = (float*)alloc(256 * 4);
  float*    bsum1  = (float*)alloc(256 * 4);
  _Float16* embA   = (_Float16*)alloc((size_t)256 * 2 * 512 * 2);   // 512 KB
  float*    Hs     = (float*)alloc((size_t)8 * 256 * 256 * 4);      // 2 MB
  _Float16* concatA= (_Float16*)alloc((size_t)128 * 16 * 512 * 2);  // 2 MB
  float*    qbuf   = (float*)alloc((size_t)2048 * 256 * 4);         // 2 MB
  float*    kTbuf  = (float*)alloc((size_t)2048 * 256 * 4);         // 2 MB
  (void)ws_size; (void)in_sizes; (void)n_in; (void)out_size;

  // weight fragment conversion
  fragB_fill<<<16384 / 256, 256, 0, stream>>>(Wih0, Wih0B, 256, 64, 2);
  fragB_fill<<<65536 / 256, 256, 0, stream>>>(Whh0, Whh0B, 256, 256, 8);
  fragB_fill<<<65536 / 256, 256, 0, stream>>>(Wih1, Wih1B, 256, 256, 8);
  fragB_fill<<<65536 / 256, 256, 0, stream>>>(Whh1, Whh1B, 256, 256, 8);
  fragB_fill<<<65536 / 256, 256, 0, stream>>>(Wq,   WqB,   256, 256, 8);
  fragB_fill<<<65536 / 256, 256, 0, stream>>>(Wk,   WkB,   256, 256, 8);
  fragB_fill<<<16384 / 256, 256, 0, stream>>>(Wfc,  WfcB,  29,  512, 16);
  bias_sum<<<1, 256, 0, stream>>>(bih0, bhh0, bih1, bhh1, bsum0, bsum1);
  embA_fill<<<262144 / 256, 256, 0, stream>>>(x, embed, embA);

  // sequential RNN: one workgroup, 16 waves
  rnn_kernel<<<1, 512, 0, stream>>>(embA, Wih0B, Whh0B, Wih1B, Whh1B,
                                    bsum0, bsum1, Hs, concatA);
  // q / k projections
  qk_kernel<<<512, 256, 0, stream>>>(concatA, WqB, WkB, qbuf, kTbuf);
  // additive attention + ctx (writes K=256..511 region of concatA)
  attn_kernel<<<2048, 256, 0, stream>>>(qbuf, kTbuf, Hs, vvec, concatA);
  // FC head
  fc_kernel<<<32, 256, 0, stream>>>(concatA, WfcB, bfc, out);
}
